// DynamicQNN_12927851561383
// MI455X (gfx1250) — matrix operand order repaired and checked by reading, not compiled, not run on hardware
//
#include <hip/hip_runtime.h>

#define NQ  10
#define DIM 1024
#define NL  4

typedef __attribute__((ext_vector_type(16))) _Float16     v16h;
typedef __attribute__((ext_vector_type(8)))  float        v8f;
typedef __attribute__((ext_vector_type(8)))  _Float16     h8;
typedef __attribute__((ext_vector_type(8)))  unsigned int u8x;

__device__ __forceinline__ int insert_bit(int x, int pos) {
    return ((x >> pos) << (pos + 1)) | (x & ((1 << pos) - 1));
}

// ---------------------------------------------------------------------------
// Kernel A: statevector simulation. ONE WAVE (32 threads) per batch element,
// statevector as float2 (re,im) in 8 KB LDS. All 512 butterfly pairs of a
// gate partition the state, so lanes never collide within a gate; between
// gates, per-wave in-order LDS execution guarantees visibility. With a
// single-wave workgroup, __syncthreads() is an S_NOP-grade compiler fence.
// ---------------------------------------------------------------------------
__global__ __launch_bounds__(32) void qsim_kernel(const float* __restrict__ x,
                                                  const float* __restrict__ wts,
                                                  _Float16* __restrict__ probs)
{
    __shared__ float2 st[DIM];          // 8 KB
    const int lane = threadIdx.x;       // 0..31
    const int b    = blockIdx.x;

    #pragma unroll
    for (int j = 0; j < DIM / 32; ++j)
        st[lane + 32 * j] = make_float2(0.f, 0.f);
    __syncthreads();
    if (lane == 0) st[0] = make_float2(1.f, 0.f);   // |0...0>
    __syncthreads();

    // ---- RX embedding: [[c, -i s],[-i s, c]], c=cos(x/2), s=sin(x/2)
    for (int q = 0; q < NQ; ++q) {
        const int bp = NQ - 1 - q;
        const int w  = 1 << bp;
        const float half = 0.5f * x[b * NQ + q];
        const float c  = cosf(half);
        const float sn = sinf(half);
        #pragma unroll
        for (int j = 0; j < 16; ++j) {
            const int p  = lane + 32 * j;                       // pair 0..511
            const int i0 = ((p >> bp) << (bp + 1)) | (p & (w - 1));
            const int i1 = i0 | w;
            const float2 a0 = st[i0];
            const float2 a1 = st[i1];
            float2 n0, n1;
            n0.x = c * a0.x + sn * a1.y;
            n0.y = c * a0.y - sn * a1.x;
            n1.x = c * a1.x + sn * a0.y;
            n1.y = c * a1.y - sn * a0.x;
            st[i0] = n0;
            st[i1] = n1;
        }
        __syncthreads();
    }

    // ---- BasicEntanglerLayers: RY (shared weights) + CNOT ring
    for (int l = 0; l < NL; ++l) {
        for (int q = 0; q < NQ; ++q) {
            const int bp = NQ - 1 - q;
            const int w  = 1 << bp;
            const float half = 0.5f * wts[l * NQ + q];
            const float ch = cosf(half);
            const float sh = sinf(half);
            #pragma unroll
            for (int j = 0; j < 16; ++j) {
                const int p  = lane + 32 * j;
                const int i0 = ((p >> bp) << (bp + 1)) | (p & (w - 1));
                const int i1 = i0 | w;
                const float2 a0 = st[i0];
                const float2 a1 = st[i1];
                float2 n0, n1;
                n0.x = ch * a0.x - sh * a1.x;
                n0.y = ch * a0.y - sh * a1.y;
                n1.x = sh * a0.x + ch * a1.x;
                n1.y = sh * a0.y + ch * a1.y;
                st[i0] = n0;
                st[i1] = n1;
            }
            __syncthreads();
        }
        // CNOT ring: control q -> target q+1 (q=0..8), then control 9 -> target 0
        for (int e = 0; e < NQ; ++e) {
            const int cb = (e < NQ - 1) ? (NQ - 1 - e) : 0;        // control bit
            const int tb = (e < NQ - 1) ? (NQ - 2 - e) : (NQ - 1); // target bit
            const int lo = cb < tb ? cb : tb;
            const int hi = cb < tb ? tb : cb;
            #pragma unroll
            for (int j = 0; j < 8; ++j) {
                const int p    = lane + 32 * j;                    // 0..255
                const int base = insert_bit(insert_bit(p, lo), hi);
                const int j0   = base | (1 << cb);  // control=1, target=0
                const int j1   = j0 | (1 << tb);    // control=1, target=1
                const float2 tmp = st[j0];
                st[j0] = st[j1];
                st[j1] = tmp;
            }
            __syncthreads();
        }
    }

    // ---- probabilities -> f16 workspace (coalesced: lane-contiguous)
    #pragma unroll
    for (int j = 0; j < DIM / 32; ++j) {
        const int d = lane + 32 * j;
        const float2 a = st[d];
        probs[(size_t)b * DIM + d] = (_Float16)(a.x * a.x + a.y * a.y);
    }
}

// ---------------------------------------------------------------------------
// Kernel B: out[B x 10] = probs[B x 1024] @ Z_SIGNS[1024 x 10] via
// v_wmma_f32_16x16x32_f16. One wave per 16-row tile, K-loop of 32 tiles.
// Sign fragment = hoisted pattern (low bits of k) XOR a per-tile packed
// sign-flip mask (high bits of k come only from kt).
// ---------------------------------------------------------------------------
__global__ __launch_bounds__(32) void zexp_wmma_kernel(const _Float16* __restrict__ probs,
                                                       float* __restrict__ out,
                                                       int B)
{
    const int lane = threadIdx.x;         // 0..31
    const int g    = lane >> 4;           // lane group
    const int n    = lane & 15;           // output column / A row-within-tile
    const int row0 = blockIdx.x * 16;

    int arow = row0 + n;
    if (arow >= B) arow = B - 1;          // read clamp (B is a multiple of 16)
    const _Float16* rowp = probs + (size_t)arow * DIM;

    // kt-invariant sign pattern: element i <-> klow = g*8 + i (i<8), 16 + g*8 + (i-8) (i>=8) (bits 0..4 of k)
    v16h pat;
    #pragma unroll
    for (int i = 0; i < 16; ++i) {
        const int klow = (i < 8) ? ((g << 3) | i) : (16 | (g << 3) | (i - 8));
        float sv;
        if (n >= NQ)       sv = 0.f;                                       // pad cols
        else if (n >= 5)   sv = ((klow >> (NQ - 1 - n)) & 1) ? -1.f : 1.f; // bit from klow
        else               sv = 1.f;                                       // bit from kt
        pat[i] = (_Float16)sv;
    }
    const u8x patu = __builtin_bit_cast(u8x, pat);
    const int  shv    = (n < 5) ? (4 - n) : 0;    // k bit (9-n) == kt bit (4-n)
    const bool hibit  = (n < 5);

    v8f acc = {};

    for (int kt = 0; kt < DIM / 32; ++kt) {
        const int kbase = kt * 32;

        // A fragment (16-bit A 16x32 layout):
        // group 0: K = {kbase+0..7, kbase+16..23}; group 1: {+8..15, +24..31}
        const h8 alo = *(const h8*)(rowp + kbase + g * 8);
        const h8 ahi = *(const h8*)(rowp + kbase + 16 + g * 8);
        v16h a;
        #pragma unroll
        for (int i = 0; i < 8; ++i) { a[i] = alo[i]; a[8 + i] = ahi[i]; }

        // B fragment: flip sign of whole pattern when kt bit (4-n) is set (n<5)
        const unsigned flip = (hibit && ((kt >> shv) & 1)) ? 0x80008000u : 0u;
        u8x bu;
        #pragma unroll
        for (int i = 0; i < 8; ++i) bu[i] = patu[i] ^ flip;
        const v16h bb = __builtin_bit_cast(v16h, bu);

        acc = __builtin_amdgcn_wmma_f32_16x16x32_f16(
            /*neg_a=*/false, a, /*neg_b=*/false, bb,
            /*c_mod=*/(short)0, acc, /*reuse_a=*/false, /*reuse_b=*/false);
    }

    // C/D layout: VGPR v <-> M = v + 8*g ; N = lane & 15
    if (row0 + 16 <= B) {                       // fast path: full tile in range
        if (n < NQ) {
            float* op = out + (size_t)row0 * NQ + n;
            #pragma unroll
            for (int v = 0; v < 8; ++v)
                op[(size_t)(v + 8 * g) * NQ] = acc[v];
        }
    } else if (n < NQ) {                        // ragged tail (unused for B=2048)
        #pragma unroll
        for (int v = 0; v < 8; ++v) {
            const int row = row0 + v + 8 * g;
            if (row < B) out[(size_t)row * NQ + n] = acc[v];
        }
    }
}

extern "C" void kernel_launch(void* const* d_in, const int* in_sizes, int n_in,
                              void* d_out, int out_size, void* d_ws, size_t ws_size,
                              hipStream_t stream) {
    const float* x   = (const float*)d_in[0];
    const float* wts = (const float*)d_in[1];
    float* out = (float*)d_out;
    _Float16* probs = (_Float16*)d_ws;   // B * 1024 halves = 4 MB for B=2048

    const int B = in_sizes[0] / NQ;      // 2048

    qsim_kernel<<<B, 32, 0, stream>>>(x, wts, probs);
    zexp_wmma_kernel<<<(B + 15) / 16, 32, 0, stream>>>(probs, out, B);
}
